// CustomRNN_37830071943883
// MI455X (gfx1250) — matrix operand order repaired and checked by reading, not compiled, not run on hardware
//
#include <hip/hip_runtime.h>
#include <hip/hip_bf16.h>
#include <math.h>

// ---------------------------------------------------------------------------
// LSTM: T=100000, I=128, H=512, gates=4H=2048, O=1
// Phase 1a/1b: pack W_ih and X into bf16 WMMA-fragment layouts (one-shot)
// Phase 1c:    x_proj = X @ W_ih^T + (b_ih+b_hh)  -- pure-load bf16 WMMA GEMM
// Phase 2:     sequential recurrence, W_hh sharded across 16 WGs' LDS (f32)
// Phase 3:     out[t] = sigmoid(logit[t] + b_out)
// ---------------------------------------------------------------------------

#define T_STEPS 100000
#define I_DIM   128
#define H_DIM   512
#define G_DIM   2048          // 4*H
#define NWG     16            // workgroups in recurrent phase
#define SLICE   32            // hidden units per WG
#define ROWS_PER_WG 128       // 4 gates * SLICE rows of W_hh per WG
#define N_TILES (T_STEPS / 16)   // 6250 row tiles
#define N_CT    (G_DIM / 16)     // 128 column tiles
#define FRAG_HALFS 2048       // halfs per tile: 4 ks * 32 lanes * 16

typedef __attribute__((ext_vector_type(16))) __bf16 v16bf;
typedef __attribute__((ext_vector_type(8)))  float  v8f;

static __device__ __forceinline__ __bf16 f2bf(float f) {
  // round-to-nearest-even f32 -> bf16
  union { float f; unsigned u; } v; v.f = f;
  unsigned r = v.u + 0x7FFFu + ((v.u >> 16) & 1u);
  unsigned short s = (unsigned short)(r >> 16);
  __bf16 b; __builtin_memcpy(&b, &s, sizeof(b));
  return b;
}

static __device__ __forceinline__ float sigmoidf(float x) {
  return 1.0f / (1.0f + expf(-x));
}

// A-fragment element -> K index (16-bit A 16x32 layout)
static __device__ __forceinline__ int a_k_of(int e, int hi) {
  int v = e >> 1, p = e & 1;
  return ((v < 4) ? (2 * v + hi * 8) : (16 + 2 * (v - 4) + hi * 8)) + p;
}
// B-fragment element -> K index (16-bit B 32x16 layout)
static __device__ __forceinline__ int b_k_of(int e, int hi) {
  return (e < 8) ? (e + hi * 8) : (16 + hi * 8 + (e - 8));
}

// ---------------------------------------------------------------------------
// Phase 0: zero logits, h double-buffer, barrier words (deterministic per call)
// ---------------------------------------------------------------------------
__global__ void lstm_init(float* __restrict__ out, float* __restrict__ hbuf,
                          unsigned* __restrict__ bar) {
  int i = blockIdx.x * blockDim.x + threadIdx.x;
  if (i < T_STEPS) out[i] = 0.0f;
  if (i < 2 * H_DIM) hbuf[i] = 0.0f;
  if (i < 2) bar[i] = 0u;
}

// ---------------------------------------------------------------------------
// Phase 1a: pack W_ih [2048,128] f32 -> Wp[ct][ks][lane][16] bf16 (512 KB).
// Fragment layout: lane n = L&15 owns column j = ct*16+n; K pairs per VGPR:
//   v<4 : K = 2v + 8*hi ; v>=4 : K = 16 + 2(v-4) + 8*hi.
// ---------------------------------------------------------------------------
__global__ __launch_bounds__(256) void pack_W(const float* __restrict__ W_ih,
                                              __bf16* __restrict__ Wp) {
  const int ct = blockIdx.x;                    // 0..127
  for (int idx = threadIdx.x; idx < FRAG_HALFS; idx += 256) {
    int ks   = idx >> 9;
    int lane = (idx >> 4) & 31;
    int e    = idx & 15;
    int n  = lane & 15, hi = lane >> 4;
    int k  = b_k_of(e, hi);
    Wp[(size_t)ct * FRAG_HALFS + idx] =
        f2bf(W_ih[(size_t)(ct * 16 + n) * I_DIM + ks * 32 + k]);
  }
}

// ---------------------------------------------------------------------------
// Phase 1b: pack X [T,128] f32 -> Xp[tile][ks][lane][16] bf16 (25.6 MB).
// Fragment layout: lane m = L&15 owns row tile*16+m; K pairs per VGPR:
//   v<4 : K = 2v + 8*hi ; v>=4 : K = 16 + 2(v-4) + 8*hi.
// ---------------------------------------------------------------------------
__global__ __launch_bounds__(256) void pack_X(const float* __restrict__ X,
                                              __bf16* __restrict__ Xp) {
  const int tile = blockIdx.x;                  // 0..6249
  for (int idx = threadIdx.x; idx < FRAG_HALFS; idx += 256) {
    int ks   = idx >> 9;
    int lane = (idx >> 4) & 31;
    int e    = idx & 15;
    int m  = lane & 15, hi = lane >> 4;
    int k  = a_k_of(e, hi);
    Xp[(size_t)tile * FRAG_HALFS + idx] =
        f2bf(X[(size_t)(tile * 16 + m) * I_DIM + ks * 32 + k]);
  }
}

// ---------------------------------------------------------------------------
// Phase 1c: x_proj[T,2048] = X @ W_ih^T + (b_ih + b_hh).
// One block = one 16-row tile; 8 waves each own 16 column tiles.
// Operands are pre-packed: each fragment is one 32-byte contiguous lane load
// (global_load_b128 x2, coalesced across the wave) feeding
// v_wmma_f32_16x16x32_bf16 directly. K-loop of 4 (I=128).
// D layout (f32 16x16): VGPR v -> row M = v + 8*hi, column N = L&15.
// ---------------------------------------------------------------------------
__global__ __launch_bounds__(256) void lstm_xproj_wmma(
    const __bf16* __restrict__ Xp,    // packed A fragments
    const __bf16* __restrict__ Wp,    // packed B fragments
    const float* __restrict__ b_ih,   // [2048]
    const float* __restrict__ b_hh,   // [2048]
    float* __restrict__ xproj)        // [T,2048]
{
  const int rowTile = blockIdx.x;          // 0..6249
  const int lane = threadIdx.x & 31;
  const int wave = threadIdx.x >> 5;       // 0..7
  const int m    = lane & 15;
  const int hi   = lane >> 4;

  // Hoist the 4 A fragments (reused across all 16 column tiles of this wave).
  const __bf16* abase = Xp + (size_t)rowTile * FRAG_HALFS + lane * 16;
  v16bf afrag[4];
#pragma unroll
  for (int ks = 0; ks < 4; ++ks)
    afrag[ks] = *reinterpret_cast<const v16bf*>(abase + ks * 512);

  for (int ct = wave; ct < N_CT; ct += 8) {
    const int j = ct * 16 + m;             // output gate column owned by lane
    const float bias = b_ih[j] + b_hh[j];
    v8f acc;
#pragma unroll
    for (int v = 0; v < 8; ++v) acc[v] = bias;

    const __bf16* bbase = Wp + (size_t)ct * FRAG_HALFS + lane * 16;
#pragma unroll
    for (int ks = 0; ks < 4; ++ks) {
      v16bf bfrag = *reinterpret_cast<const v16bf*>(bbase + ks * 512);
      acc = __builtin_amdgcn_wmma_f32_16x16x32_bf16(
          false, afrag[ks], false, bfrag, (short)0, acc, false, false);
    }

#pragma unroll
    for (int v = 0; v < 8; ++v) {
      int r = rowTile * 16 + v + hi * 8;
      xproj[(size_t)r * G_DIM + j] = acc[v];
    }
  }
}

// ---------------------------------------------------------------------------
// Phase 2: sequential LSTM recurrence. 16 persistent WGs; WG g owns hidden
// slice [g*32, g*32+32) and the 128 corresponding W_hh rows in LDS (f32,
// 256 KB of the 320 KB WGP LDS). Per step: LDS mat-vec, cell update, publish
// h-slice to the double-buffered global h, wave-reduce partial readout into
// out[t] via global f32 atomic, then cluster barrier (fast path when launched
// as a 16-WG cluster; NOP otherwise) + device-scope atomic sense barrier.
// ---------------------------------------------------------------------------
__global__ __launch_bounds__(256) void lstm_recur(
    const float* __restrict__ W_hh,   // [2048,512]
    const float* __restrict__ W_out,  // [512]
    const float* __restrict__ xproj,  // [T,2048]
    float* __restrict__ hbuf,         // [2,512]
    unsigned* __restrict__ bar,       // [2]: arrive count, generation
    float* __restrict__ out)          // [T] logit accumulator (pre-zeroed)
{
  extern __shared__ float smem[];
  float* wsl  = smem;                          // 128*512 f32 weights
  float* hcur = wsl + ROWS_PER_WG * H_DIM;     // 512
  float* red  = hcur + H_DIM;                  // 256
  float* garr = red + 256;                     // 128

  const int wg  = blockIdx.x;                  // 0..15
  const int tid = threadIdx.x;                 // 0..255

  // Preload this WG's 128 gate rows of W_hh (i/f/g/o interleave by 512).
  for (int idx = tid; idx < ROWS_PER_WG * H_DIM; idx += 256) {
    int lr = idx / H_DIM;
    int k  = idx - lr * H_DIM;
    int gr = (lr >> 5) * H_DIM + wg * SLICE + (lr & 31);
    wsl[idx] = W_hh[(size_t)gr * H_DIM + k];
  }
  __syncthreads();

  float c_reg = 0.0f;                                        // tid<32 owns c
  const float wout = (tid < SLICE) ? W_out[wg * SLICE + tid] : 0.0f;

  const int r    = tid & 127;                                // local gate row
  const int half = tid >> 7;                                 // K halves
  const int grow = (r >> 5) * H_DIM + wg * SLICE + (r & 31); // global gate row

  for (int t = 0; t < T_STEPS; ++t) {
    // Stage current h into LDS.
    const float* hsrc = hbuf + (t & 1) * H_DIM;
    for (int i = tid; i < H_DIM; i += 256) hcur[i] = hsrc[i];
    __syncthreads();

    // Prefetch next step's x_proj element (global_prefetch_b8).
    if (half == 0 && t + 1 < T_STEPS)
      __builtin_prefetch(&xproj[(size_t)(t + 1) * G_DIM + grow], 0, 0);

    // Mat-vec partial: each thread does 256 f32 FMAs from LDS.
    const float* wrow = wsl + (size_t)r * H_DIM + half * 256;
    const float* hv   = hcur + half * 256;
    float acc = (half == 0) ? xproj[(size_t)t * G_DIM + grow] : 0.0f;
#pragma unroll 8
    for (int k = 0; k < 256; ++k) acc = fmaf(wrow[k], hv[k], acc);
    red[tid] = acc;
    __syncthreads();
    if (tid < ROWS_PER_WG) garr[tid] = red[tid] + red[tid + 128];
    __syncthreads();

    // Cell update on lanes 0..31 of wave 0.
    if (tid < SLICE) {
      float iv = sigmoidf(garr[tid]);
      float fv = sigmoidf(garr[SLICE + tid]);
      float gv = tanhf(garr[2 * SLICE + tid]);
      float ov = sigmoidf(garr[3 * SLICE + tid]);
      c_reg = fv * c_reg + iv * gv;
      float hval = ov * tanhf(c_reg);
      hbuf[((t + 1) & 1) * H_DIM + wg * SLICE + tid] = hval;

      // Partial readout dot, wave32 xor-reduce, one atomic per WG per step.
      float pd = hval * wout;
#pragma unroll
      for (int off = 16; off > 0; off >>= 1)
        pd += __shfl_xor(pd, off, 32);
      if (tid == 0) atomicAdd(&out[t], pd);
    }

    // Make h-slice + logit visible device-wide before the step barrier.
    __threadfence();
    __syncthreads();

    // Cluster barrier: real sync when dispatched as a 16-WG cluster,
    // architecturally a NOP otherwise (ClusterID==0 -> S_NOP).
    __builtin_amdgcn_s_cluster_barrier();

    // Device-scope sense barrier across the 16 WGs (correct in all cases).
    if (tid == 0) {
      unsigned gen = __hip_atomic_load(&bar[1], __ATOMIC_RELAXED,
                                       __HIP_MEMORY_SCOPE_AGENT);
      unsigned arr = __hip_atomic_fetch_add(&bar[0], 1u, __ATOMIC_ACQ_REL,
                                            __HIP_MEMORY_SCOPE_AGENT);
      if (arr == NWG - 1) {
        __hip_atomic_store(&bar[0], 0u, __ATOMIC_RELAXED,
                           __HIP_MEMORY_SCOPE_AGENT);
        __hip_atomic_fetch_add(&bar[1], 1u, __ATOMIC_RELEASE,
                               __HIP_MEMORY_SCOPE_AGENT);
      } else {
        while (__hip_atomic_load(&bar[1], __ATOMIC_ACQUIRE,
                                 __HIP_MEMORY_SCOPE_AGENT) == gen)
          __builtin_amdgcn_s_sleep(1);
      }
    }
    __syncthreads();
  }
}

// ---------------------------------------------------------------------------
// Phase 3: out[t] = sigmoid(logit[t] + b_out)
// ---------------------------------------------------------------------------
__global__ void lstm_out_sigmoid(float* __restrict__ out,
                                 const float* __restrict__ b_out) {
  int t = blockIdx.x * blockDim.x + threadIdx.x;
  if (t < T_STEPS) out[t] = sigmoidf(out[t] + b_out[0]);
}

// ---------------------------------------------------------------------------
extern "C" void kernel_launch(void* const* d_in, const int* in_sizes, int n_in,
                              void* d_out, int out_size, void* d_ws, size_t ws_size,
                              hipStream_t stream) {
  (void)in_sizes; (void)n_in; (void)out_size; (void)ws_size;

  const float* X     = (const float*)d_in[0];  // [T,128]
  const float* W_ih  = (const float*)d_in[1];  // [2048,128]
  const float* b_ih  = (const float*)d_in[2];  // [2048]
  const float* W_hh  = (const float*)d_in[3];  // [2048,512]
  const float* b_hh  = (const float*)d_in[4];  // [2048]
  const float* W_out = (const float*)d_in[5];  // [1,512]
  const float* b_out = (const float*)d_in[6];  // [1]
  float* out = (float*)d_out;                  // [T]

  char* ws = (char*)d_ws;
  size_t off = 0;
  float* xproj = (float*)(ws + off);                       // T*2048 f32
  off += (size_t)T_STEPS * G_DIM * sizeof(float);
  float* hbuf = (float*)(ws + off);                        // 2*512 f32
  off += 2 * H_DIM * sizeof(float);
  unsigned* bar = (unsigned*)(ws + off);                   // 2 u32 (+pad)
  off += 256;
  __bf16* Xp = (__bf16*)(ws + off);                        // T*128 bf16
  off += (size_t)N_TILES * FRAG_HALFS * sizeof(__bf16);
  __bf16* Wp = (__bf16*)(ws + off);                        // 2048*128 bf16
  off += (size_t)N_CT * FRAG_HALFS * sizeof(__bf16);

  lstm_init<<<(T_STEPS + 255) / 256, 256, 0, stream>>>(out, hbuf, bar);

  pack_W<<<N_CT, 256, 0, stream>>>(W_ih, Wp);
  pack_X<<<N_TILES, 256, 0, stream>>>(X, Xp);

  lstm_xproj_wmma<<<N_TILES, 256, 0, stream>>>(Xp, Wp, b_ih, b_hh, xproj);

  size_t smem = (size_t)(ROWS_PER_WG * H_DIM + H_DIM + 256 + 128) * sizeof(float);
  lstm_recur<<<NWG, 256, smem, stream>>>(W_hh, W_out, xproj, hbuf, bar, out);

  lstm_out_sigmoid<<<(T_STEPS + 255) / 256, 256, 0, stream>>>(out, b_out);
}
